// StockElmanCell_2396591751748
// MI455X (gfx1250) — matrix operand order repaired and checked by reading, not compiled, not run on hardware
//
#include <hip/hip_runtime.h>
#include <hip/hip_bf16.h>

// ---------------- problem constants (fixed by reference) ----------------
#define T_STEPS 2048
#define BATCH   16
#define DIM     1024
#define TBD     (2048LL * 16LL * 1024LL)   // rows of the big GEMM * D
#define ROWS    (2048 * 16)                // 32768 (t,b) rows
#define NWG_REC 32                         // persistent workgroups for recurrence

typedef __attribute__((ext_vector_type(16))) __bf16 v16bf;
typedef __attribute__((ext_vector_type(8)))  float  v8f;
typedef __attribute__((ext_vector_type(4))) unsigned int u32x4;
typedef __attribute__((ext_vector_type(8)))  int i32x8;
typedef __attribute__((ext_vector_type(4)))  int i32x4;

// ---------------- helpers ----------------
__device__ __forceinline__ unsigned short f2bf(float f) {
    unsigned u = __float_as_uint(f);
    unsigned r = u + 0x7FFFu + ((u >> 16) & 1u);   // round-to-nearest-even
    return (unsigned short)(r >> 16);
}
__device__ __forceinline__ float bf2f(unsigned short h) {
    return __uint_as_float(((unsigned)h) << 16);
}

union FragU { uint4 q[2]; v16bf v; };

// A-fragment: 16x32 bf16 tile, row-major source [lda], rows m0..m0+15, k0..k0+31.
// Layout: lane<16 -> M=lane, K=0..15 contiguous;
//         lane>=16 -> same M, K=16..31 (matches B fragment order).
__device__ __forceinline__ v16bf load_a_frag(const unsigned short* __restrict__ A,
                                             int lda, int m0, int k0, int lane) {
    const unsigned short* p = A + (size_t)(m0 + (lane & 15)) * lda + k0 + ((lane >> 4) << 4);
    FragU u;
    u.q[0] = *(const uint4*)(p);        // K base..base+7
    u.q[1] = *(const uint4*)(p + 8);    // K base+8..base+15
    return u.v;
}

// B-fragment: 32x16 bf16 tile where B[k][n] = W[n][k] (W row-major [ldw]).
// Layout used here (contiguous k, same as A): lanes 0-15 hold K=0..15 of column N=lane; lanes 16-31 hold K=16..31.
__device__ __forceinline__ v16bf load_b_frag(const unsigned short* __restrict__ W,
                                             int ldw, int n0, int k0, int lane) {
    const unsigned short* p = W + (size_t)(n0 + (lane & 15)) * ldw + k0 + ((lane >> 4) << 4);
    FragU u;
    u.q[0] = *(const uint4*)(p);
    u.q[1] = *(const uint4*)(p + 8);
    return u.v;
}

// Same fragment gather but from an LDS-resident panel [32 rows][1024 cols].
__device__ __forceinline__ v16bf load_b_frag_lds(const unsigned short* P,
                                                 int n0loc, int k0, int lane) {
    const unsigned short* p = P + (n0loc + (lane & 15)) * DIM + k0 + ((lane >> 4) << 4);
    FragU u;
    u.q[0] = *(const uint4*)(p);
    u.q[1] = *(const uint4*)(p + 8);
    return u.v;
}

__device__ __forceinline__ v8f wmma_bf16(v16bf a, v16bf b, v8f c) {
    return __builtin_amdgcn_wmma_f32_16x16x32_bf16(false, a, false, b, (short)0, c, false, false);
}

// ---------------- TDM: async 2-D bf16 tile load  global -> LDS ----------------
// Packs a Tensor DMA Descriptor (D#) per CDNA5 ISA 08_async_tensor §8:
//   group0: count=1 | lds_addr | global_addr[56:0] | type=2
//   group1: data_size=2B | tensor dims (large) | tile_dim0=cols, tile_dim1=rows
//           | tensor_dim0_stride = row_stride (elements)
// groups 2/3 zero (2-D tensor). Tracked by TENSORcnt.
__device__ __forceinline__ void tdm_load_2d_bf16(const unsigned short* gsrc,
                                                 unsigned lds_addr_bytes,
                                                 int rows, int cols, int row_stride) {
    unsigned long long ga = (unsigned long long)gsrc;
    u32x4 g0;
    g0[0] = 1u;                                            // count=1, user mode
    g0[1] = lds_addr_bytes;                                // lds_addr
    g0[2] = (unsigned)(ga & 0xFFFFFFFFu);                  // global_addr[31:0]
    g0[3] = (unsigned)((ga >> 32) & 0x01FFFFFFu) | (2u << 30); // addr[56:32] | type=2

    const unsigned td0 = 1u << 30;                         // huge tensor dims: tile in-bounds
    const unsigned td1 = 1u << 30;
    i32x8 g1;
    g1[0] = (int)(1u << 16);                               // data_size=1 (2 bytes), mask=0
    g1[1] = (int)((td0 & 0xFFFFu) << 16);                  // tensor_dim0[15:0] @ bits 79:48
    g1[2] = (int)((td0 >> 16) | ((td1 & 0xFFFFu) << 16));  // tensor_dim0[31:16] | tensor_dim1[15:0]
    g1[3] = (int)((td1 >> 16) | ((unsigned)cols << 16));   // tensor_dim1[31:16] | tile_dim0
    g1[4] = (int)((unsigned)rows & 0xFFFFu);               // tile_dim1 (tile_dim2=0)
    g1[5] = (int)(unsigned)row_stride;                     // tensor_dim0_stride[31:0]
    g1[6] = 0;                                             // stride[47:32]=0 | dim1_stride lo=0
    g1[7] = 0;

    i32x4 z4 = {0, 0, 0, 0};
#if defined(__clang_major__) && (__clang_major__ >= 23)
    i32x8 z8 = {0, 0, 0, 0, 0, 0, 0, 0};
    __builtin_amdgcn_tensor_load_to_lds(g0, g1, z4, z4, z8, 0);
#else
    __builtin_amdgcn_tensor_load_to_lds(g0, g1, z4, z4, 0);
#endif
}

// ---------------- kernel 0: convert inputs to bf16, seed state ----------------
__global__ void __launch_bounds__(256)
prep_kernel(const float* __restrict__ x, const float* __restrict__ h0,
            const float* __restrict__ Wx, const float* __restrict__ Wh,
            const float* __restrict__ Wg,
            unsigned short* __restrict__ xbf, unsigned short* __restrict__ wxbf,
            unsigned short* __restrict__ whbf, unsigned short* __restrict__ wgbf,
            unsigned short* __restrict__ h_ws, float* __restrict__ h_all0,
            unsigned* __restrict__ ctr) {
    long long i = (long long)blockIdx.x * blockDim.x + threadIdx.x;
    if (i < TBD)               xbf[i] = f2bf(x[i]);
    if (i < DIM * DIM) {
        wxbf[i] = f2bf(Wx[i]);
        whbf[i] = f2bf(Wh[i]);
        wgbf[i] = f2bf(Wg[i]);
    }
    if (i < BATCH * DIM) {
        h_ws[i]   = f2bf(h0[i]);
        h_all0[i] = h0[i];                 // h_all[0] = h0
    }
    if (i == 0) *ctr = 0u;                 // reset grid barrier each launch
}

// ---------------- kernel 1: fused dual GEMM  pre_x / silu-gate ----------------
// Block = 8 waves sharing one 32-column weight panel; TDM stages the full
// 32x1024 bf16 panels of W_x and W_gate into LDS (128 KB). K-loop is manually
// software-pipelined: iteration k issues loads for k+32, consumes fragments
// loaded one iteration earlier, so s_wait before WMMA covers already-retired
// loads instead of stalling.
__global__ void __launch_bounds__(256)
gemm_xw_kernel(const unsigned short* __restrict__ xbf,
               const unsigned short* __restrict__ Wx,
               const unsigned short* __restrict__ Wg,
               const float* __restrict__ bias,
               const float* __restrict__ bias_g,
               float* __restrict__ pre_x,
               unsigned short* __restrict__ gate) {
    extern __shared__ unsigned short ldsB[];          // [2][32][1024] bf16 = 128 KB
    unsigned short* panelX = ldsB;
    unsigned short* panelG = ldsB + 32 * DIM;

    const int lane = threadIdx.x & 31;
    const int wave = threadIdx.x >> 5;
    const int nt   = blockIdx.x >> 8;                 // 32 n-tiles of 32 cols
    const int mt   = ((blockIdx.x & 255) << 3) + wave;// 2048 m-tiles, 8 per block
    const int m0   = mt << 4;
    const int n0   = nt << 5;

    if (threadIdx.x < 32) {                           // wave 0 issues both DMAs
        unsigned lds0 = (unsigned)(unsigned long long)(const void*)ldsB;
        tdm_load_2d_bf16(Wx + (size_t)n0 * DIM, lds0,                32, DIM, DIM);
        tdm_load_2d_bf16(Wg + (size_t)n0 * DIM, lds0 + 32 * DIM * 2, 32, DIM, DIM);
        __builtin_amdgcn_s_wait_tensorcnt(0);
    }
    __syncthreads();

    // ---- software pipeline: prologue loads for k = 0 ----
    v16bf a_c  = load_a_frag(xbf, DIM, m0, 0, lane);
    v16bf bx0c = load_b_frag_lds(panelX, 0,  0, lane);
    v16bf bx1c = load_b_frag_lds(panelX, 16, 0, lane);
    v16bf bg0c = load_b_frag_lds(panelG, 0,  0, lane);
    v16bf bg1c = load_b_frag_lds(panelG, 16, 0, lane);

    v8f ax0 = {}, ax1 = {}, ag0 = {}, ag1 = {};
    #pragma unroll 4
    for (int k = 0; k < DIM; k += 32) {
        v16bf a_n, bx0n, bx1n, bg0n, bg1n;
        const int kn = k + 32;
        if (kn < DIM) {                               // issue next iteration's loads first
            a_n  = load_a_frag(xbf, DIM, m0, kn, lane);
            bx0n = load_b_frag_lds(panelX, 0,  kn, lane);
            bx1n = load_b_frag_lds(panelX, 16, kn, lane);
            bg0n = load_b_frag_lds(panelG, 0,  kn, lane);
            bg1n = load_b_frag_lds(panelG, 16, kn, lane);
            __builtin_prefetch(xbf + (size_t)(m0 + (lane & 15)) * DIM + kn, 0, 1);
        }
        ax0 = wmma_bf16(a_c, bx0c, ax0);              // consume previous iteration's frags
        ax1 = wmma_bf16(a_c, bx1c, ax1);
        ag0 = wmma_bf16(a_c, bg0c, ag0);
        ag1 = wmma_bf16(a_c, bg1c, ag1);
        a_c = a_n; bx0c = bx0n; bx1c = bx1n; bg0c = bg0n; bg1c = bg1n;
    }

    const int hi    = (lane >> 4) & 1;                // C layout: M = v + 8*hi, N = lane&15
    const int rbase = m0 + (hi << 3);
    const int nloc  = lane & 15;
    const float bx0 = bias[n0 + nloc],   bx1 = bias[n0 + 16 + nloc];
    const float bg0 = bias_g[n0 + nloc], bg1 = bias_g[n0 + 16 + nloc];
    #pragma unroll
    for (int v = 0; v < 8; ++v) {
        size_t r = (size_t)(rbase + v) * DIM;
        float p0 = ax0[v] + bx0;
        float p1 = ax1[v] + bx1;
        pre_x[r + n0 + nloc]      = p0;
        pre_x[r + n0 + 16 + nloc] = p1;
        float gv0 = ag0[v] + bg0;
        float gv1 = ag1[v] + bg1;
        gate[r + n0 + nloc]       = f2bf(gv0 / (1.0f + __expf(-gv0)));
        gate[r + n0 + 16 + nloc]  = f2bf(gv1 / (1.0f + __expf(-gv1)));
    }
}

// ---------------- kernel 2: persistent sequential recurrence ----------------
// 32 WGs x 256 threads; WG owns 32 output columns. W_h slice lives in VGPRs as
// pre-built B-fragments for the whole scan. One grid barrier per timestep.
__global__ void __launch_bounds__(256)
recurrence_kernel(const float* __restrict__ pre_x,
                  const unsigned short* __restrict__ gate,
                  const unsigned short* __restrict__ Wh,
                  unsigned short* __restrict__ h_ws,
                  float* __restrict__ outs,
                  float* __restrict__ h_all,
                  unsigned* __restrict__ ctr) {
    __shared__ float lds[8 * 2 * 256];        // 16 KB of per-wave partial tiles
    const int tid  = threadIdx.x;
    const int lane = tid & 31;
    const int w    = tid >> 5;                // wave 0..7 -> K slice of 128
    const int n_wg = blockIdx.x << 5;         // 32 columns per WG

    // Pin the W_h slice in registers: 8 B-fragments (4 k-steps x 2 n-tiles).
    v16bf bw[4][2];
    #pragma unroll
    for (int kk = 0; kk < 4; ++kk) {
        #pragma unroll
        for (int nti = 0; nti < 2; ++nti)
            bw[kk][nti] = load_b_frag(Wh, DIM, n_wg + nti * 16, w * 128 + kk * 32, lane);
    }

    volatile unsigned* vctr = (volatile unsigned*)ctr;
    const unsigned nwg = gridDim.x;

    for (int t = 0; t < T_STEPS; ++t) {
        // Hoist all 4 A-fragment loads ahead of the 8 WMMAs: one wait per step.
        v16bf a0 = load_a_frag(h_ws, DIM, 0, w * 128 + 0 * 32, lane);
        v16bf a1 = load_a_frag(h_ws, DIM, 0, w * 128 + 1 * 32, lane);
        v16bf a2 = load_a_frag(h_ws, DIM, 0, w * 128 + 2 * 32, lane);
        v16bf a3 = load_a_frag(h_ws, DIM, 0, w * 128 + 3 * 32, lane);
        v8f acc0 = {}, acc1 = {};
        acc0 = wmma_bf16(a0, bw[0][0], acc0);  acc1 = wmma_bf16(a0, bw[0][1], acc1);
        acc0 = wmma_bf16(a1, bw[1][0], acc0);  acc1 = wmma_bf16(a1, bw[1][1], acc1);
        acc0 = wmma_bf16(a2, bw[2][0], acc0);  acc1 = wmma_bf16(a2, bw[2][1], acc1);
        acc0 = wmma_bf16(a3, bw[3][0], acc0);  acc1 = wmma_bf16(a3, bw[3][1], acc1);

        #pragma unroll
        for (int v = 0; v < 8; ++v) {
            lds[(w * 2 + 0) * 256 + v * 32 + lane] = acc0[v];
            lds[(w * 2 + 1) * 256 + v * 32 + lane] = acc1[v];
        }
        __syncthreads();

        // reduce 8 waves, finalize 2 of the 512 tile elements per thread
        #pragma unroll
        for (int j = 0; j < 2; ++j) {
            int idx = tid * 2 + j;
            int m   = idx & 15;               // batch row
            int nl  = idx >> 4;               // 0..31 local column
            int nti = nl >> 4;
            int n   = nl & 15;
            int sl  = n + ((m >> 3) << 4);    // C-layout lane of element (m,n)
            float s = 0.0f;
            #pragma unroll
            for (int ww = 0; ww < 8; ++ww)
                s += lds[(ww * 2 + nti) * 256 + (m & 7) * 32 + sl];
            int    col = n_wg + nl;
            size_t ro  = (size_t)(t * BATCH + m) * DIM + col;
            float  hn  = tanhf(pre_x[ro] + s);
            outs[ro]                                         = hn * bf2f(gate[ro]);
            h_all[(size_t)((t + 1) * BATCH + m) * DIM + col] = hn;
            h_ws[m * DIM + col]                              = f2bf(hn);
        }
        __threadfence();                      // release this WG's h slice
        __syncthreads();

        // device-wide phase barrier (persistent grid, one arrival per WG)
        if (tid == 0) {
            atomicAdd(ctr, 1u);
            unsigned target = nwg * (unsigned)(t + 1);
            while (*vctr < target) { __builtin_amdgcn_s_sleep(2); }
        }
        __syncthreads();
        __threadfence();                      // acquire: invalidate WGP$ before re-reading h
    }
}

// ---------------- host-side launch ----------------
extern "C" void kernel_launch(void* const* d_in, const int* in_sizes, int n_in,
                              void* d_out, int out_size, void* d_ws, size_t ws_size,
                              hipStream_t stream) {
    (void)in_sizes; (void)n_in; (void)out_size; (void)ws_size;
    const float* x   = (const float*)d_in[0];
    const float* h0  = (const float*)d_in[1];
    const float* Wx  = (const float*)d_in[2];
    const float* Wh  = (const float*)d_in[3];
    const float* Wg  = (const float*)d_in[4];
    const float* b   = (const float*)d_in[5];
    const float* bg  = (const float*)d_in[6];

    float* outs  = (float*)d_out;                       // [T, B, D]
    float* h_all = outs + TBD;                          // [T+1, B, D]

    // workspace carve-up (all offsets 256B aligned)
    char* ws = (char*)d_ws;
    unsigned short* xbf  = (unsigned short*)ws;                 ws += TBD * 2;          // 64 MB
    unsigned short* wxbf = (unsigned short*)ws;                 ws += DIM * DIM * 2;    // 2 MB
    unsigned short* whbf = (unsigned short*)ws;                 ws += DIM * DIM * 2;    // 2 MB
    unsigned short* wgbf = (unsigned short*)ws;                 ws += DIM * DIM * 2;    // 2 MB
    unsigned short* h_ws = (unsigned short*)ws;                 ws += BATCH * DIM * 2;  // 32 KB
    float*          prex = (float*)ws;                          ws += TBD * 4;          // 128 MB
    unsigned short* gate = (unsigned short*)ws;                 ws += TBD * 2;          // 64 MB
    unsigned*       ctr  = (unsigned*)ws;

    // 0) convert + seed
    {
        long long n = TBD;
        int blocks = (int)((n + 255) / 256);
        prep_kernel<<<blocks, 256, 0, stream>>>(x, h0, Wx, Wh, Wg,
                                                xbf, wxbf, whbf, wgbf,
                                                h_ws, h_all /* row 0 */, ctr);
    }
    // 1) big fused GEMM: 8 waves/block share a TDM-staged 128 KB weight panel
    {
        int blocks = (ROWS / 16) / 8 * (DIM / 32);   // 256 * 32 = 8192
        size_t shmem = 2u * 32u * DIM * sizeof(unsigned short);  // 128 KB
        gemm_xw_kernel<<<blocks, 256, shmem, stream>>>(xbf, wxbf, wgbf, b, bg, prex, gate);
    }
    // 2) sequential scan: 32 persistent WGs
    {
        recurrence_kernel<<<NWG_REC, 256, 0, stream>>>(prex, gate, whbf, h_ws,
                                                       outs, h_all, ctr);
    }
}
